// RSSMTransition_68539088109569
// MI455X (gfx1250) — matrix operand order repaired and checked by reading, not compiled, not run on hardware
//
#include <hip/hip_runtime.h>
#include <hip/hip_bf16.h>

// ---------------------------------------------------------------------------
// RSSM transition, MI455X (gfx1250), bf16 WMMA (f32 accumulate) implementation.
// Shapes: B=65536, A=6, S=30, D=H=200, C=32. K padded: 36->64, 200->224.
// N padded: 200->208, 600->608, 960->960 (already x16).
// ---------------------------------------------------------------------------

typedef __bf16 bf16;
typedef __attribute__((ext_vector_type(16))) __bf16 v16bf;
typedef __attribute__((ext_vector_type(8)))  __bf16 v8bf;
typedef __attribute__((ext_vector_type(8)))  float  v8f;

#define LN_EPS 1e-5f

// ------------------------- wave32 reductions -------------------------------
__device__ __forceinline__ float wred_sum(float v) {
#pragma unroll
  for (int m = 16; m >= 1; m >>= 1) v += __shfl_xor(v, m, 32);
  return v;
}
// Reductions across the 16 lanes of each half-wave (xor masks stay in-half).
__device__ __forceinline__ float hred_sum(float v) {
#pragma unroll
  for (int m = 8; m >= 1; m >>= 1) v += __shfl_xor(v, m, 32);
  return v;
}
__device__ __forceinline__ float hred_max(float v) {
#pragma unroll
  for (int m = 8; m >= 1; m >>= 1) v = fmaxf(v, __shfl_xor(v, m, 32));
  return v;
}

// ------------------------- WMMA fragment helpers ---------------------------
// A fragment (16x32 bf16, rows = batch): contiguous-k order (same as k_pack).
// lane l: m = l&15, half = l>>4; values k = half*16 + 0..15.
__device__ __forceinline__ v16bf load_a_frag(const bf16* __restrict__ base, int ldk) {
  const int lane = threadIdx.x & 31;
  const int m  = lane & 15;
  const int kh = lane >> 4;
  const bf16* p = base + (size_t)m * ldk + kh * 16;
  v8bf lo = *(const v8bf*)(p);
  v8bf hi = *(const v8bf*)(p + 8);
  v16bf a;
#pragma unroll
  for (int i = 0; i < 8; ++i) { a[i] = lo[i]; a[i + 8] = hi[i]; }
  return a;
}

// B fragment: packed tile is 512 bf16; lane l owns 16 contiguous bf16.
__device__ __forceinline__ v16bf load_b_frag(const bf16* __restrict__ tile) {
  const int lane = threadIdx.x & 31;
  return *(const v16bf*)(tile + lane * 16);
}

__device__ __forceinline__ v8f wmma_bf16(v16bf a, v16bf b, v8f c) {
  return __builtin_amdgcn_wmma_f32_16x16x32_bf16(false, a, false, b,
                                                 (short)0, c, false, false);
}

__device__ __forceinline__ float eluf(float x) {
  return x > 0.f ? x : (__expf(x) - 1.f);
}
__device__ __forceinline__ float sigm(float x) {
  return 1.f / (1.f + __expf(-x));
}

// ---------------------------------------------------------------------------
// Pack W [N][K] fp32 row-major -> bf16 tiles [KT][NT][512] in B-frag layout:
// within tile: elem (lane,j): n = nt*16 + (lane&15), k = kt*32 + (lane>>4)*16 + j
// ---------------------------------------------------------------------------
__global__ void k_pack(const float* __restrict__ W, bf16* __restrict__ out,
                       int K, int N, int KT, int NT) {
  int tid = blockIdx.x * blockDim.x + threadIdx.x;
  int total = KT * NT * 512;
  if (tid >= total) return;
  int tile = tid >> 9;
  int within = tid & 511;
  int lane = within >> 4;
  int j = within & 15;
  int kt = tile / NT;
  int nt = tile - kt * NT;
  int n = nt * 16 + (lane & 15);
  int k = kt * 32 + (lane >> 4) * 16 + j;
  float v = (k < K && n < N) ? W[(size_t)n * K + k] : 0.f;
  out[tid] = (bf16)v;
}

// Stage concat(prev_action, prev_stoch) -> bf16 [B][64] (zero padded)
__global__ void k_stage_x(const float* __restrict__ act,
                          const float* __restrict__ stoch,
                          bf16* __restrict__ xbf, int B) {
  int tid = blockIdx.x * blockDim.x + threadIdx.x;
  if (tid >= B * 64) return;
  int b = tid >> 6, k = tid & 63;
  float v = 0.f;
  if (k < 6) v = act[b * 6 + k];
  else if (k < 36) v = stoch[b * 30 + (k - 6)];
  xbf[tid] = (bf16)v;
}

// Stage fp32 [B][K] -> bf16 [B][Kp] (zero padded)
__global__ void k_stage_pad(const float* __restrict__ src, bf16* __restrict__ dst,
                            int B, int K, int Kp) {
  int tid = blockIdx.x * blockDim.x + threadIdx.x;
  if (tid >= B * Kp) return;
  int b = tid / Kp, k = tid - b * Kp;
  dst[tid] = (bf16)((k < K) ? src[(size_t)b * K + k] : 0.f);
}

// ---------------------------------------------------------------------------
// K2: rnn_in = elu(x @ W_in^T + b_in) -> bf16 [B][224]. K=64 (2 ktiles),
// Np=208 (13 ntiles); also writes zero pad cols 200..223 (nt loop to 14).
// ---------------------------------------------------------------------------
__global__ void __launch_bounds__(256) k_gemm_in(
    const bf16* __restrict__ xbf, const bf16* __restrict__ pk,
    const float* __restrict__ b_in, bf16* __restrict__ rnnbf) {
  const int lane = threadIdx.x & 31;
  const int wave = threadIdx.x >> 5;
  const int row0 = (blockIdx.x * (blockDim.x >> 5) + wave) * 16;
  const v16bf a0 = load_a_frag(xbf + (size_t)row0 * 64 +  0, 64);
  const v16bf a1 = load_a_frag(xbf + (size_t)row0 * 64 + 32, 64);
  const int colLane = lane & 15;
  const int rowHalf = (lane >> 4) * 8;
  for (int nt = 0; nt < 14; ++nt) {
    v8f acc = {};
    if (nt < 13) {
      acc = wmma_bf16(a0, load_b_frag(pk + (size_t)(0 * 13 + nt) * 512), acc);
      acc = wmma_bf16(a1, load_b_frag(pk + (size_t)(1 * 13 + nt) * 512), acc);
    }
    const int col = nt * 16 + colLane;
    const float bias = (col < 200) ? b_in[col] : 0.f;
#pragma unroll
    for (int r = 0; r < 8; ++r) {
      const int row = row0 + rowHalf + r;
      float v = (col < 200) ? eluf(acc[r] + bias) : 0.f;
      rnnbf[(size_t)row * 224 + col] = (bf16)v;
    }
  }
}

// ---------------------------------------------------------------------------
// K3: GRU. gi = rnn_in @ W_ih^T + b_ih ; gh = prev_det @ W_hh^T + b_hh
// (both [16 rows][608] fp32 in per-wave LDS slabs), gates, LayerNorm -> det.
// Gate mixing crosses tiles at +200/+400 (not 16-aligned) so LDS is the right
// tool here. B-fragment loads are double-buffered to overlap VMEM with WMMA.
// 2 waves / block, LDS = 2*2*16*608*4 = 155,648 B.
// ---------------------------------------------------------------------------
__global__ void __launch_bounds__(64) k_gru(
    const bf16* __restrict__ rnnbf, const bf16* __restrict__ dprevbf,
    const bf16* __restrict__ pkih, const bf16* __restrict__ pkhh,
    const float* __restrict__ b_ih, const float* __restrict__ b_hh,
    const float* __restrict__ prev_det,
    const float* __restrict__ ln_g, const float* __restrict__ ln_b,
    float* __restrict__ det_out, bf16* __restrict__ detbf) {
  extern __shared__ float sm[];
  const int lane = threadIdx.x & 31;
  const int wave = threadIdx.x >> 5;
  float* gi = sm + (size_t)wave * 2 * 16 * 608;
  float* gh = gi + 16 * 608;
  const int row0 = (blockIdx.x * (blockDim.x >> 5) + wave) * 16;

  v16bf ai[7], ah[7];
#pragma unroll
  for (int kt = 0; kt < 7; ++kt) {
    ai[kt] = load_a_frag(rnnbf  + (size_t)row0 * 224 + kt * 32, 224);
    ah[kt] = load_a_frag(dprevbf + (size_t)row0 * 224 + kt * 32, 224);
  }
  const int colLane = lane & 15;
  const int rowHalf = (lane >> 4) * 8;
  for (int nt = 0; nt < 38; ++nt) {
    v8f acc_i = {}, acc_h = {};
    v16bf bi = load_b_frag(pkih + (size_t)(0 * 38 + nt) * 512);
    v16bf bh = load_b_frag(pkhh + (size_t)(0 * 38 + nt) * 512);
#pragma unroll
    for (int kt = 0; kt < 7; ++kt) {
      v16bf biN = bi, bhN = bh;
      if (kt < 6) {  // issue next tile's loads before consuming current
        biN = load_b_frag(pkih + (size_t)((kt + 1) * 38 + nt) * 512);
        bhN = load_b_frag(pkhh + (size_t)((kt + 1) * 38 + nt) * 512);
      }
      acc_i = wmma_bf16(ai[kt], bi, acc_i);
      acc_h = wmma_bf16(ah[kt], bh, acc_h);
      bi = biN; bh = bhN;
    }
    const int col = nt * 16 + colLane;
    const float bib = (col < 600) ? b_ih[col] : 0.f;
    const float bhb = (col < 600) ? b_hh[col] : 0.f;
#pragma unroll
    for (int r = 0; r < 8; ++r) {
      const int rr = rowHalf + r;
      gi[rr * 608 + col] = acc_i[r] + bib;
      gh[rr * 608 + col] = acc_h[r] + bhb;
    }
  }
  // gates + layernorm (per row, lanes parallel over cols)
  for (int rr = 0; rr < 16; ++rr) {
    const int row = row0 + rr;
    float sum = 0.f, sumsq = 0.f;
    for (int c = lane; c < 200; c += 32) {
      const float r = sigm(gi[rr * 608 + c]       + gh[rr * 608 + c]);
      const float z = sigm(gi[rr * 608 + c + 200] + gh[rr * 608 + c + 200]);
      const float n = tanhf(gi[rr * 608 + c + 400] + r * gh[rr * 608 + c + 400]);
      const float pd = prev_det[(size_t)row * 200 + c];
      const float d = (1.f - z) * n + z * pd;
      gi[rr * 608 + c] = d;  // stash
      sum += d; sumsq += d * d;
    }
    sum = wred_sum(sum); sumsq = wred_sum(sumsq);
    const float mean = sum * (1.f / 200.f);
    const float var  = sumsq * (1.f / 200.f) - mean * mean;
    const float inv  = rsqrtf(var + LN_EPS);
    for (int c = lane; c < 224; c += 32) {
      float y = 0.f;
      if (c < 200) {
        y = (gi[rr * 608 + c] - mean) * inv * ln_g[c] + ln_b[c];
        det_out[(size_t)row * 200 + c] = y;
      }
      detbf[(size_t)row * 224 + c] = (bf16)y;
    }
  }
}

// ---------------------------------------------------------------------------
// K5: h = LN(elu(det @ W1^T + b1)) -> bf16 [B][224].  LDS-free: the 13 output
// tiles (104 VGPRs/lane) stay in registers; LN stats via half-wave reductions
// (a row of the D matrix lives across the 16 lanes of one half-wave).
// ---------------------------------------------------------------------------
__global__ void __launch_bounds__(256) k_mlp1(
    const bf16* __restrict__ detbf, const bf16* __restrict__ pkw1,
    const float* __restrict__ b1,
    const float* __restrict__ ln_g, const float* __restrict__ ln_b,
    bf16* __restrict__ hbf) {
  const int lane = threadIdx.x & 31;
  const int wave = threadIdx.x >> 5;
  const int row0 = (blockIdx.x * (blockDim.x >> 5) + wave) * 16;

  v16bf a[7];
#pragma unroll
  for (int kt = 0; kt < 7; ++kt)
    a[kt] = load_a_frag(detbf + (size_t)row0 * 224 + kt * 32, 224);

  const int colLane = lane & 15;
  const int rowHalf = (lane >> 4) * 8;
  float h[13][8];
#pragma unroll
  for (int nt = 0; nt < 13; ++nt) {
    v8f acc = {};
    v16bf b = load_b_frag(pkw1 + (size_t)(0 * 13 + nt) * 512);
#pragma unroll
    for (int kt = 0; kt < 7; ++kt) {
      v16bf bN = b;
      if (kt < 6) bN = load_b_frag(pkw1 + (size_t)((kt + 1) * 13 + nt) * 512);
      acc = wmma_bf16(a[kt], b, acc);
      b = bN;
    }
    const int col = nt * 16 + colLane;
    const float bias = (col < 200) ? b1[col] : 0.f;
#pragma unroll
    for (int r = 0; r < 8; ++r)
      h[nt][r] = (col < 200) ? eluf(acc[r] + bias) : 0.f;
  }
#pragma unroll
  for (int r = 0; r < 8; ++r) {
    float s = 0.f, q = 0.f;
#pragma unroll
    for (int nt = 0; nt < 13; ++nt) { s += h[nt][r]; q += h[nt][r] * h[nt][r]; }
    s = hred_sum(s); q = hred_sum(q);
    const float mean = s * (1.f / 200.f);
    const float var  = q * (1.f / 200.f) - mean * mean;
    const float inv  = rsqrtf(var + LN_EPS);
    const int row = row0 + rowHalf + r;
#pragma unroll
    for (int nt = 0; nt < 13; ++nt) {
      const int col = nt * 16 + colLane;
      float y = 0.f;
      if (col < 200) y = (h[nt][r] - mean) * inv * ln_g[col] + ln_b[col];
      hbf[(size_t)row * 224 + col] = (bf16)y;
    }
    hbf[(size_t)row * 224 + 208 + colLane] = (bf16)0.f;  // pad cols 208..223
  }
}

// ---------------------------------------------------------------------------
// K6: logits = h @ W2^T + b2, softmax + unimix + stats, fully in registers.
// Each softmax group (C=32 cols) == exactly two 16x16 WMMA tiles; a logits row
// lives across one half-wave (lane&15 = col, lane>>4 = row half), so max/sum/
// var reduce with xor masks {1,2,4,8}. No LDS at all.
// NOTE: sample.mean(C) == 1/C exactly (one-hot), so stoch pre-LN is constant
// -> LN output is just ln_s_b[s]; mean = 1/C exactly (computed anyway).
// ---------------------------------------------------------------------------
__global__ void __launch_bounds__(256) k_logits(
    const bf16* __restrict__ hbf, const bf16* __restrict__ pkw2,
    const float* __restrict__ b2, const float* __restrict__ ln_s_b,
    float* __restrict__ out_mean, float* __restrict__ out_std,
    float* __restrict__ out_stoch) {
  const int lane = threadIdx.x & 31;
  const int wave = threadIdx.x >> 5;
  const int row0 = (blockIdx.x * (blockDim.x >> 5) + wave) * 16;

  v16bf a[7];
#pragma unroll
  for (int kt = 0; kt < 7; ++kt)
    a[kt] = load_a_frag(hbf + (size_t)row0 * 224 + kt * 32, 224);

  const int colLane = lane & 15;
  const int rowHalf = (lane >> 4) * 8;
  for (int s = 0; s < 30; ++s) {
    if (s + 1 < 30)  // stream next group's W2 tiles through L2
      __builtin_prefetch(pkw2 + (size_t)(2 * s + 2) * 512, 0, 1);
    v8f acc0 = {}, acc1 = {};
    v16bf b0 = load_b_frag(pkw2 + (size_t)(0 * 60 + 2 * s)     * 512);
    v16bf b1 = load_b_frag(pkw2 + (size_t)(0 * 60 + 2 * s + 1) * 512);
#pragma unroll
    for (int kt = 0; kt < 7; ++kt) {
      v16bf b0N = b0, b1N = b1;
      if (kt < 6) {
        b0N = load_b_frag(pkw2 + (size_t)((kt + 1) * 60 + 2 * s)     * 512);
        b1N = load_b_frag(pkw2 + (size_t)((kt + 1) * 60 + 2 * s + 1) * 512);
      }
      acc0 = wmma_bf16(a[kt], b0, acc0);
      acc1 = wmma_bf16(a[kt], b1, acc1);
      b0 = b0N; b1 = b1N;
    }
    const float bias0 = b2[s * 32 + colLane];
    const float bias1 = b2[s * 32 + 16 + colLane];
    const float stoch_v = ln_s_b[s];
#pragma unroll
    for (int r = 0; r < 8; ++r) {
      const float v0 = acc0[r] + bias0;
      const float v1 = acc1[r] + bias1;
      const float mx = hred_max(fmaxf(v0, v1));
      const float e0 = __expf(v0 - mx), e1 = __expf(v1 - mx);
      const float den = hred_sum(e0 + e1);
      const float sc = 0.99f / den;
      const float p0 = e0 * sc + (0.01f / 32.f);
      const float p1 = e1 * sc + (0.01f / 32.f);
      const float pm = hred_sum(p0 + p1) * (1.f / 32.f);  // == 1/32 exactly
      const float d0 = p0 - pm, d1 = p1 - pm;
      const float var = hred_sum(d0 * d0 + d1 * d1) * (1.f / 31.f);  // ddof=1
      if (colLane == 0) {
        const int row = row0 + rowHalf + r;
        const size_t o = (size_t)row * 30 + s;
        out_mean[o]  = pm;
        out_std[o]   = sqrtf(var);
        out_stoch[o] = stoch_v;  // LN of constant vector -> beta
      }
    }
  }
}

// ---------------------------------------------------------------------------
extern "C" void kernel_launch(void* const* d_in, const int* in_sizes, int n_in,
                              void* d_out, int out_size, void* d_ws, size_t ws_size,
                              hipStream_t stream) {
  const float* prev_action = (const float*)d_in[0];
  const float* prev_stoch  = (const float*)d_in[1];
  const float* prev_det    = (const float*)d_in[2];
  const float* W_in = (const float*)d_in[3];
  const float* b_in = (const float*)d_in[4];
  const float* W_ih = (const float*)d_in[5];
  const float* W_hh = (const float*)d_in[6];
  const float* b_ih = (const float*)d_in[7];
  const float* b_hh = (const float*)d_in[8];
  const float* ln_det_g = (const float*)d_in[9];
  const float* ln_det_b = (const float*)d_in[10];
  const float* W1 = (const float*)d_in[11];
  const float* b1 = (const float*)d_in[12];
  const float* ln_h_g = (const float*)d_in[13];
  const float* ln_h_b = (const float*)d_in[14];
  const float* W2 = (const float*)d_in[15];
  const float* b2 = (const float*)d_in[16];
  const float* ln_s_b = (const float*)d_in[18];  // ln_s_g multiplied by 0, unused

  const int B = in_sizes[0] / 6;  // 65536

  // Workspace carve-out (256B-aligned regions)
  char* ws = (char*)d_ws;
  size_t off = 0;
  auto carve = [&](size_t bytes) -> void* {
    void* p = ws + off;
    off += (bytes + 255) & ~(size_t)255;
    return p;
  };
  bf16* pk_in = (bf16*)carve((size_t)64 * 208 * 2);     // KT=2,  NT=13
  bf16* pk_ih = (bf16*)carve((size_t)224 * 608 * 2);    // KT=7,  NT=38
  bf16* pk_hh = (bf16*)carve((size_t)224 * 608 * 2);
  bf16* pk_w1 = (bf16*)carve((size_t)224 * 208 * 2);    // KT=7,  NT=13
  bf16* pk_w2 = (bf16*)carve((size_t)224 * 960 * 2);    // KT=7,  NT=60
  bf16* xbf     = (bf16*)carve((size_t)B * 64 * 2);
  bf16* rnnbf   = (bf16*)carve((size_t)B * 224 * 2);
  bf16* dprevbf = (bf16*)carve((size_t)B * 224 * 2);
  bf16* detbf   = (bf16*)carve((size_t)B * 224 * 2);
  bf16* hbf     = (bf16*)carve((size_t)B * 224 * 2);
  (void)ws_size;

  float* out_mean  = (float*)d_out;
  float* out_std   = out_mean + (size_t)B * 30;
  float* out_stoch = out_mean + (size_t)2 * B * 30;
  float* out_det   = out_mean + (size_t)3 * B * 30;

  auto g = [](int total) { return dim3((total + 255) / 256); };

  // Weight packing (small)
  k_pack<<<g(2 * 13 * 512), 256, 0, stream>>>(W_in, pk_in, 36, 200, 2, 13);
  k_pack<<<g(7 * 38 * 512), 256, 0, stream>>>(W_ih, pk_ih, 200, 600, 7, 38);
  k_pack<<<g(7 * 38 * 512), 256, 0, stream>>>(W_hh, pk_hh, 200, 600, 7, 38);
  k_pack<<<g(7 * 13 * 512), 256, 0, stream>>>(W1,  pk_w1, 200, 200, 7, 13);
  k_pack<<<g(7 * 60 * 512), 256, 0, stream>>>(W2,  pk_w2, 200, 960, 7, 60);

  // Activation staging
  k_stage_x<<<g(B * 64), 256, 0, stream>>>(prev_action, prev_stoch, xbf, B);
  k_stage_pad<<<g(B * 224), 256, 0, stream>>>(prev_det, dprevbf, B, 200, 224);

  // Fused WMMA pipeline
  k_gemm_in<<<dim3(B / 128), 256, 0, stream>>>(xbf, pk_in, b_in, rnnbf);
  k_gru<<<dim3(B / 32), 64, 2 * 2 * 16 * 608 * sizeof(float), stream>>>(
      rnnbf, dprevbf, pk_ih, pk_hh, b_ih, b_hh, prev_det,
      ln_det_g, ln_det_b, out_det, detbf);
  k_mlp1<<<dim3(B / 128), 256, 0, stream>>>(detbf, pk_w1, b1, ln_h_g, ln_h_b, hbf);
  k_logits<<<dim3(B / 128), 256, 0, stream>>>(hbf, pk_w2, b2, ln_s_b,
                                              out_mean, out_std, out_stoch);
}
